// LSTMCell_60988535603585
// MI455X (gfx1250) — matrix operand order repaired and checked by reading, not compiled, not run on hardware
//
#include <hip/hip_runtime.h>

typedef __attribute__((ext_vector_type(16))) _Float16 v16h;
typedef __attribute__((ext_vector_type(8)))  _Float16 v8h;
typedef __attribute__((ext_vector_type(8)))  float    v8f;
typedef __attribute__((ext_vector_type(4)))  float    f32x4;
typedef __attribute__((ext_vector_type(2)))  unsigned int u32x2;
typedef __attribute__((ext_vector_type(4)))  unsigned int u32x4;
typedef __attribute__((ext_vector_type(4)))  int          i32x4;
typedef __attribute__((ext_vector_type(8)))  int          i32x8;

#define B_ROWS   131072
#define IN_DIM   128
#define H_DIM    128
#define K_DIM    256                        // IN + H (concatenated GEMM K)
#define NG       512                        // 4 * H gate columns
#define NBLOCKS  512
#define ROWS_PER_BLOCK (B_ROWS / NBLOCKS)   // 256
#define M_ITER   32                         // rows per inner iteration (2 WMMA row tiles)
#define N_ITERS  (ROWS_PER_BLOCK / M_ITER)  // 8
#define THREADS  256                        // 8 wave32
#define A_HALVES (M_ITER * K_DIM)           // halves per A buffer

// LDS: f16 weights [NG][K_DIM] + double-buffered A tile 2x[M_ITER][K_DIM]
#define SMEM_HALVES (NG * K_DIM + 2 * A_HALVES)
#define SMEM_BYTES  (SMEM_HALVES * 2)       // 294912 B = 288 KB (<= 320 KB WGP LDS)

// ---------------------------------------------------------------------------
// Prep: pack Wx (512x128) | Wh (512x128) into f16 Wc[col][k] (contiguous K per
// gate column = WMMA B-operand layout).
// ---------------------------------------------------------------------------
__global__ void lstm_prep_f16(const float* __restrict__ Wx,
                              const float* __restrict__ Wh,
                              _Float16* __restrict__ Wc) {
  int idx = blockIdx.x * blockDim.x + threadIdx.x;
  if (idx < NG * K_DIM) {
    int col = idx >> 8;
    int k   = idx & (K_DIM - 1);
    float v = (k < IN_DIM) ? Wx[col * IN_DIM + k] : Wh[col * H_DIM + (k - IN_DIM)];
    Wc[idx] = (_Float16)v;
  }
}

// CDNA5 native transcendental tanh (TRANS op; v_nop covers the TRANS hazard
// before the result is consumed / sources overwritten).
__device__ __forceinline__ float fast_tanh(float x) {
  float r;
  asm("v_tanh_f32 %0, %1\n\tv_nop" : "=v"(r) : "v"(x));
  return r;
}
__device__ __forceinline__ float fast_sigmoid(float x) {
  return __builtin_fmaf(0.5f, fast_tanh(0.5f * x), 0.5f);
}

// Stage one A tile: M_ITER rows x K_DIM f16, K = [x | h_prev], via NT loads +
// packed RTZ converts.
__device__ __forceinline__ void stage_tile(const float* __restrict__ x,
                                           const float* __restrict__ h_prev,
                                           _Float16* __restrict__ dst,
                                           int row0, int tid) {
  for (int i = tid; i < (M_ITER * K_DIM) / 4; i += THREADS) {
    const int r  = i >> 6;           // row in tile (64 quads per row)
    const int k4 = (i & 63) << 2;    // K quad base (never straddles 128)
    const float* src = (k4 < IN_DIM)
        ? (x      + (size_t)(row0 + r) * IN_DIM + k4)
        : (h_prev + (size_t)(row0 + r) * H_DIM  + (k4 - IN_DIM));
    f32x4 v = __builtin_nontemporal_load((const f32x4*)src);
    u32x2 pk;
    pk[0] = __builtin_bit_cast(unsigned int, __builtin_amdgcn_cvt_pkrtz(v[0], v[1]));
    pk[1] = __builtin_bit_cast(unsigned int, __builtin_amdgcn_cvt_pkrtz(v[2], v[3]));
    *(u32x2*)(dst + r * K_DIM + k4) = pk;
  }
}

// ---------------------------------------------------------------------------
// Main fused LSTM cell: TDM weight stage + double-buffered A + f16 WMMA.
// ---------------------------------------------------------------------------
__global__ __launch_bounds__(THREADS)
void lstm_cell_wmma(const float* __restrict__ x,
                    const float* __restrict__ h_prev,
                    const float* __restrict__ c_prev,
                    const _Float16* __restrict__ Wc,
                    const float* __restrict__ bx,
                    const float* __restrict__ bh,
                    float* __restrict__ h_out,
                    float* __restrict__ c_out) {
  extern __shared__ _Float16 smem[];
  _Float16* Wl    = smem;               // [NG][K_DIM], LDS offset 0
  _Float16* Abase = smem + NG * K_DIM;  // 2 x [M_ITER][K_DIM] double buffer

  const int tid  = threadIdx.x;
  const int wave = tid >> 5;           // 0..7 -> owns output cols [16w, 16w+16)
  const int lane = tid & 31;
  const int nloc = lane & 15;          // N (and M) index within a 16x16 tile
  const int sel  = lane >> 4;          // lane-half selector for fragment layout

  // ---- Stage f16 weights into LDS (256 KB) via Tensor Data Mover ----
#if __has_builtin(__builtin_amdgcn_tensor_load_to_lds) && __has_builtin(__builtin_amdgcn_s_wait_tensorcnt)
  if (wave == 0) {
    const unsigned long long ga = (unsigned long long)(uintptr_t)Wc;
    u32x4 g0;
    g0[0] = 1u;                                    // count=1 (valid user D#)
    g0[1] = 0u;                                    // lds_addr = 0 (Wl at LDS base)
    g0[2] = (unsigned int)ga;                      // global_addr[31:0]
    g0[3] = (unsigned int)((ga >> 32) & 0x1FFFFFFu) | (2u << 30); // addr[56:32] | type=2
    i32x8 g1 = {};
    g1[0] = (1 << 16);                             // data_size = 1 (2 bytes)
    g1[1] = (int)((unsigned)K_DIM << 16);          // tensor_dim0 = 256 (bits 79:48)
    g1[2] = (int)((unsigned)NG << 16);             // tensor_dim1 = 512 (bits 111:80)
    g1[3] = (int)((unsigned)K_DIM << 16);          // tile_dim0 = 256 (bits 127:112)
    g1[4] = NG;                                    // tile_dim1 = 512, tile_dim2 = 0
    g1[5] = K_DIM;                                 // tensor_dim0_stride = 256
    const i32x4 gz = {};
#if defined(__clang_major__) && __clang_major__ >= 23
    const i32x8 gz8 = {};
    __builtin_amdgcn_tensor_load_to_lds(g0, g1, gz, gz, gz8, 0);
#else
    __builtin_amdgcn_tensor_load_to_lds(g0, g1, gz, gz, 0);
#endif
    __builtin_amdgcn_s_wait_tensorcnt(0);
  }
#else
  {
    const u32x4* src = (const u32x4*)Wc;
    u32x4* dst = (u32x4*)Wl;
    for (int i = tid; i < (NG * K_DIM) / 8; i += THREADS) dst[i] = src[i];
  }
#endif

  // ---- Per-lane fused biases for the 4 gates of this lane's output column ----
  const int ncol = wave * 16 + nloc;   // 0..127
  const float bias_i = bx[ncol]       + bh[ncol];
  const float bias_f = bx[ncol + 128] + bh[ncol + 128];
  const float bias_g = bx[ncol + 256] + bh[ncol + 256];
  const float bias_o = bx[ncol + 384] + bh[ncol + 384];

  const int row_base = blockIdx.x * ROWS_PER_BLOCK;

  // Prologue: stage first A tile (overlaps with wave 0's TDM wait)
  stage_tile(x, h_prev, Abase, row_base, tid);

  for (int it = 0; it < N_ITERS; ++it) {
    __syncthreads();   // tile 'it' (and weights on it==0) visible; buffer reuse safe
    const _Float16* Al = Abase + (it & 1) * A_HALVES;
    const int row0 = row_base + it * M_ITER;

    // Issue next tile's global loads early: they overlap this tile's WMMAs.
    if (it + 1 < N_ITERS)
      stage_tile(x, h_prev, Abase + ((it + 1) & 1) * A_HALVES, row0 + M_ITER, tid);

    // Preload c_prev so its latency hides under the K loop.
    float cp[2][8];
#pragma unroll
    for (int rt = 0; rt < 2; ++rt)
#pragma unroll
      for (int v = 0; v < 8; ++v)
        cp[rt][v] = __builtin_nontemporal_load(
            c_prev + (size_t)(row0 + rt * 16 + v + sel * 8) * H_DIM + ncol);

    // ---- K loop: 8 steps of 16x16x32 f16 WMMA, 2 row tiles x 4 gates ----
    v8f acc[2][4];
#pragma unroll
    for (int rt = 0; rt < 2; ++rt)
#pragma unroll
      for (int g = 0; g < 4; ++g) acc[rt][g] = (v8f)(0.0f);

#pragma unroll
    for (int kk = 0; kk < K_DIM / 32; ++kk) {
      const int k0 = kk * 32;

      // A fragments: 16-bit A 16x32 layout -> lane (M=nloc) reads 8 halves at
      // k0 + sel*8 and 8 halves at k0 + 16 + sel*8.
      v16h a[2];
#pragma unroll
      for (int rt = 0; rt < 2; ++rt) {
        const _Float16* ab = Al + (rt * 16 + nloc) * K_DIM + k0 + sel * 8;
        v8h lo = *(const v8h*)(ab);
        v8h hi = *(const v8h*)(ab + 16);
#pragma unroll
        for (int e = 0; e < 8; ++e) { a[rt][e] = lo[e]; a[rt][8 + e] = hi[e]; }
      }

      // B fragments per gate: 16-bit B 32x16 layout -> lane (N=nloc) reads 8
      // halves at k0 + sel*8 and 8 halves at k0 + 16 + sel*8; reused across both row tiles.
#pragma unroll
      for (int g = 0; g < 4; ++g) {
        const _Float16* bb = Wl + (size_t)(g * 128 + ncol) * K_DIM + k0 + sel * 8;
        v8h blo = *(const v8h*)(bb);
        v8h bhi = *(const v8h*)(bb + 16);
        v16h bfrag;
#pragma unroll
        for (int e = 0; e < 8; ++e) { bfrag[e] = blo[e]; bfrag[8 + e] = bhi[e]; }

        acc[0][g] = __builtin_amdgcn_wmma_f32_16x16x32_f16(
            false, a[0], false, bfrag, (short)0, acc[0][g], false, false);
        acc[1][g] = __builtin_amdgcn_wmma_f32_16x16x32_f16(
            false, a[1], false, bfrag, (short)0, acc[1][g], false, false);
      }
    }

    // ---- Epilogue: fused gates -> (h, c) via native v_tanh_f32 ----
    // C/D layout: acc[rt][g][v] is row (rt*16 + v + 8*sel), col ncol.
#pragma unroll
    for (int rt = 0; rt < 2; ++rt) {
#pragma unroll
      for (int v = 0; v < 8; ++v) {
        const int row = row0 + rt * 16 + v + sel * 8;
        const size_t off = (size_t)row * H_DIM + ncol;
        const float gi = fast_sigmoid(acc[rt][0][v] + bias_i);
        const float gf = fast_sigmoid(acc[rt][1][v] + bias_f);
        const float gg = fast_tanh   (acc[rt][2][v] + bias_g);
        const float go = fast_sigmoid(acc[rt][3][v] + bias_o);
        const float c  = gf * cp[rt][v] + gi * gg;
        __builtin_nontemporal_store(go * fast_tanh(c), h_out + off);
        __builtin_nontemporal_store(c, c_out + off);
      }
    }
  }
}

// ---------------------------------------------------------------------------
// Host launcher. Inputs (setup_inputs order): x, h_prev, c_prev, Wx, bx, Wh, bh.
// Output: h (B*H floats) then c (B*H floats), concatenated.
// Workspace: 256 KB packed f16 weights.
// ---------------------------------------------------------------------------
extern "C" void kernel_launch(void* const* d_in, const int* in_sizes, int n_in,
                              void* d_out, int out_size, void* d_ws, size_t ws_size,
                              hipStream_t stream) {
  const float* x      = (const float*)d_in[0];
  const float* h_prev = (const float*)d_in[1];
  const float* c_prev = (const float*)d_in[2];
  const float* Wx     = (const float*)d_in[3];
  const float* bx     = (const float*)d_in[4];
  const float* Wh     = (const float*)d_in[5];
  const float* bh     = (const float*)d_in[6];

  float* h_out = (float*)d_out;
  float* c_out = h_out + (size_t)B_ROWS * H_DIM;

  _Float16* Wc = (_Float16*)d_ws;   // NG*K_DIM f16 = 256 KB

  (void)hipFuncSetAttribute((const void*)lstm_cell_wmma,
                            hipFuncAttributeMaxDynamicSharedMemorySize,
                            (int)SMEM_BYTES);

  lstm_prep_f16<<<(NG * K_DIM + 255) / 256, 256, 0, stream>>>(Wx, Wh, Wc);

  lstm_cell_wmma<<<NBLOCKS, THREADS, SMEM_BYTES, stream>>>(
      x, h_prev, c_prev, Wc, bx, bh, h_out, c_out);
}
